// SwinTransformerBlock_kv_39599598469822
// MI455X (gfx1250) — matrix operand order repaired and checked by reading, not compiled, not run on hardware
//
#include <hip/hip_runtime.h>
#include <hip/hip_bf16.h>
#include <math.h>

typedef _Float16 f16;
typedef __attribute__((ext_vector_type(16))) _Float16 v16h;
typedef __attribute__((ext_vector_type(8)))  float    v8f;

#define WMMA_F32_F16(a, b, c) \
  __builtin_amdgcn_wmma_f32_16x16x32_f16(false, (a), false, (b), (short)0, (c), false, false)

// ---------------------------------------------------------------------------
// Problem constants
// ---------------------------------------------------------------------------
#define C_DIM   192
#define HD_DIM  768          // 4*C
#define NH      6
#define HD      32           // C/NH
#define WS      4
#define N_WIN   64           // WS^3
#define NWIN    512          // windows per batch (8^3)
#define B_BAT   2
#define ROWS    65536        // B * L = B_ * N
#define L_TOK   32768
#define SHIFT   2

// ---------------------------------------------------------------------------
// WMMA operand helpers.
// A tile 16x32 (MxK): lanes 0-15 -> rows; elems 0-15 -> K{0..15 | 16..31}
// by lane half.
// B tile 32x16 (KxN): lane -> column, elems -> contiguous K in lane half.
// C/D: vgpr j holds row j (lanes 0-15) / j+8 (lanes 16-31), col = lane&15.
// ---------------------------------------------------------------------------
__device__ inline v16h load_A32(const f16* __restrict__ base, int stride, int lane) {
  int m = lane & 15;
  int koff = (lane & 16) ? 16 : 0;
  const f16* row = base + m * stride + koff;
  v16h a;
#pragma unroll
  for (int i = 0; i < 8; ++i) a[i] = row[i];
#pragma unroll
  for (int i = 0; i < 8; ++i) a[i + 8] = row[8 + i];
  return a;
}

__device__ inline v8f v8f_zero() {
  v8f z;
#pragma unroll
  for (int i = 0; i < 8; ++i) z[i] = 0.0f;
  return z;
}

// Issue the async global->LDS copies for one 64x32 f16 A tile (2 x B128/thread).
__device__ inline void stage_tileA(const f16* __restrict__ A, long lda, long rowBase,
                                   int kt, f16* dst, int tid) {
#pragma unroll
  for (int u = 0; u < 2; ++u) {
    int c = tid + u * 128;
    int row = c >> 2, ch = c & 3;
    unsigned lds = (unsigned)(unsigned long long)(dst + row * 32 + ch * 8);
    unsigned long long g =
        (unsigned long long)(A + (rowBase + row) * lda + kt + ch * 8);
    asm volatile("global_load_async_to_lds_b128 %0, %1, off"
                 :: "v"(lds), "v"(g) : "memory");
  }
}

// ---------------------------------------------------------------------------
// GEMM core: 128 threads (4 waves). Block computes 64 rows x 64 cols of
// A(MxK f16 row-major) @ B, B pre-swizzled into WMMA operand tiles:
// Bsw[((kt*NT + nt)*32 + lane)*16 + e]. Wave w owns column tile ntIdx.
// A tiles are double-buffered in LDS via async B128 copies (ASYNCcnt):
// tile t+1 streams in while tile t is consumed by the WMMAs.
// ---------------------------------------------------------------------------
__device__ inline void gemm_tile_sw(const f16* __restrict__ A, long lda, long rowBase,
                                    const f16* __restrict__ Bsw, int NT, int ntIdx,
                                    int K, v8f acc[4], f16* ldsA /* 2*64*32 */) {
  const int tid  = threadIdx.x;
  const int lane = tid & 31;
#pragma unroll
  for (int m = 0; m < 4; ++m) acc[m] = v8f_zero();

  const int KT = K >> 5;
  stage_tileA(A, lda, rowBase, 0, ldsA, tid);
  for (int t = 0; t < KT; ++t) {
    // This wave's tile-t copies are the only outstanding async ops.
    asm volatile("s_wait_asynccnt 0x0" ::: "memory");
    __syncthreads();  // all waves staged tile t; all reads of buf[(t+1)&1] done
    f16* cur = ldsA + (t & 1) * (64 * 32);
    if (t + 1 < KT)
      stage_tileA(A, lda, rowBase, (t + 1) * 32, ldsA + ((t + 1) & 1) * (64 * 32), tid);

    // B operand: one contiguous 32B/lane read from pre-swizzled weights
    const f16* bp = Bsw + (((long)t * NT + ntIdx) * 32 + lane) * 16;
    v16h b = *(const v16h*)bp;
#pragma unroll
    for (int mt = 0; mt < 4; ++mt) {
      v16h a = load_A32(cur + mt * 16 * 32, 32, lane);
      acc[mt] = WMMA_F32_F16(a, b, acc[mt]);
    }
  }
}

// Window-layout row -> (b, l) of the shifted source/destination token.
__device__ inline long win_row_to_l(long r, int* bOut) {
  int b  = (int)(r >> 15);
  int wi = (int)((r >> 6) & 511);
  int n  = (int)(r & 63);
  int wz = wi >> 6, wy = (wi >> 3) & 7, wx = wi & 7;
  int iz = n >> 4, iy = (n >> 2) & 3, ix = n & 3;
  int z = (wz * WS + iz + SHIFT) & 31;
  int y = (wy * WS + iy + SHIFT) & 31;
  int x = (wx * WS + ix + SHIFT) & 31;
  *bOut = b;
  return ((long)z * 32 + y) * 32 + x;
}

// ---------------------------------------------------------------------------
// K0: weight conversion + swizzle into B-operand tile order; rpb gather
// ---------------------------------------------------------------------------
__device__ inline void swz(const float* __restrict__ W, f16* __restrict__ out,
                           long idx, int NT, int N) {
  int within = (int)(idx & 511);
  long tile = idx >> 9;
  int lane = within >> 4, e = within & 15;
  int kt = (int)(tile / NT), nt = (int)(tile % NT);
  int n = nt * 16 + (lane & 15);
  int k = kt * 32 + ((lane & 16) ? 16 : 0) + e;
  out[idx] = (f16)W[(long)k * N + n];
}

__global__ void k_prep(const float* __restrict__ kv_w, const float* __restrict__ proj_w,
                       const float* __restrict__ fc1_w, const float* __restrict__ fc2_w,
                       const float* __restrict__ rpb_table,
                       f16* wKV, f16* wPROJ, f16* wFC1, f16* wFC2, float* rpbF) {
  long i = (long)blockIdx.x * 256 + threadIdx.x;
  if (i < (long)C_DIM * 2 * C_DIM) swz(kv_w,   wKV,   i, 24, 2 * C_DIM);
  if (i < (long)C_DIM * C_DIM)     swz(proj_w, wPROJ, i, 12, C_DIM);
  if (i < (long)C_DIM * HD_DIM)    swz(fc1_w,  wFC1,  i, 48, HD_DIM);
  if (i < (long)HD_DIM * C_DIM)    swz(fc2_w,  wFC2,  i, 12, C_DIM);
  if (i < NH * N_WIN * N_WIN) {
    int h = (int)(i >> 12);
    int q = (int)((i >> 6) & 63);
    int k = (int)(i & 63);
    int qz = q >> 4, qy = (q >> 2) & 3, qx = q & 3;
    int kz = k >> 4, ky = (k >> 2) & 3, kx = k & 3;
    int idx = (qz - kz + WS - 1) * (3 * WS - 1)
            + (qy - ky + WS - 1) * (2 * WS - 1)
            + (qx - kx + WS - 1);
    rpbF[i] = rpb_table[idx * NH + h];
  }
}

// ---------------------------------------------------------------------------
// K1: LayerNorm1 + cyclic shift + window partition (skip -> skw, x_up -> qw)
// ---------------------------------------------------------------------------
__global__ void k_ln_part(const float* __restrict__ skip, const float* __restrict__ xup,
                          const float* __restrict__ g, const float* __restrict__ bb,
                          f16* __restrict__ skw, f16* __restrict__ qw) {
  long r = (long)blockIdx.x * 8 + (threadIdx.x >> 5);
  int lane = threadIdx.x & 31;
  int b;
  long l = win_row_to_l(r, &b);
  long src = ((long)b * L_TOK + l) * C_DIM;
  const float* in0 = skip + src;
  const float* in1 = xup + src;
  f16* out0 = skw + r * C_DIM;
  f16* out1 = qw + r * C_DIM;
#pragma unroll
  for (int t = 0; t < 2; ++t) {
    const float* in = t ? in1 : in0;
    f16* out = t ? out1 : out0;
    float v[6], s = 0.0f;
#pragma unroll
    for (int j = 0; j < 6; ++j) { v[j] = in[lane + 32 * j]; s += v[j]; }
#pragma unroll
    for (int o = 16; o >= 1; o >>= 1) s += __shfl_xor(s, o, 32);
    float mean = s * (1.0f / C_DIM);
    float vs = 0.0f;
#pragma unroll
    for (int j = 0; j < 6; ++j) { float d = v[j] - mean; vs += d * d; }
#pragma unroll
    for (int o = 16; o >= 1; o >>= 1) vs += __shfl_xor(vs, o, 32);
    float inv = rsqrtf(vs * (1.0f / C_DIM) + 1e-5f);
#pragma unroll
    for (int j = 0; j < 6; ++j) {
      int c = lane + 32 * j;
      out[c] = (f16)((v[j] - mean) * inv * g[c] + bb[c]);
    }
  }
}

// ---------------------------------------------------------------------------
// K2: KV GEMM. K-half -> kkh[row][192] (key-major). V-half -> vT, transposed
// per (window, head): vT[((win*NH + h)*32 + d)*64 + key].
// ---------------------------------------------------------------------------
__global__ void k_gemm_kv(const f16* __restrict__ A, const f16* __restrict__ Bsw,
                          const float* __restrict__ bias,
                          f16* __restrict__ kkh, f16* __restrict__ vT) {
  __shared__ f16 lA[2 * 64 * 32];
  v8f acc[4];
  long rowBase = (long)blockIdx.x * 64;
  int ntIdx = blockIdx.y * 4 + (threadIdx.x >> 5);
  gemm_tile_sw(A, C_DIM, rowBase, Bsw, 24, ntIdx, C_DIM, acc, lA);
  int lane = threadIdx.x & 31;
  int n = ntIdx * 16 + (lane & 15);
  float bv = bias[n];
  int mo = (lane & 16) ? 8 : 0;
  if (n < C_DIM) {
#pragma unroll
    for (int mt = 0; mt < 4; ++mt)
#pragma unroll
      for (int j = 0; j < 8; ++j) {
        long m = rowBase + mt * 16 + j + mo;
        kkh[m * C_DIM + n] = (f16)(acc[mt][j] + bv);
      }
  } else {
    int d = n - C_DIM;
    int hh = d >> 5, dd = d & 31;
#pragma unroll
    for (int mt = 0; mt < 4; ++mt)
#pragma unroll
      for (int j = 0; j < 8; ++j) {
        long m = rowBase + mt * 16 + j + mo;
        long win = m >> 6;
        int key = (int)(m & 63);
        vT[((win * NH + hh) * 32 + dd) * 64 + key] = (f16)(acc[mt][j] + bv);
      }
  }
}

// ---------------------------------------------------------------------------
// K3: fused per-(window, head) attention, 1 wave / block.
// Q/K/V operands loaded directly from global (contiguous 32B per lane);
// only the softmax relayout uses LDS.
// ---------------------------------------------------------------------------
__global__ void k_attn(const f16* __restrict__ qw, const f16* __restrict__ kkh,
                       const f16* __restrict__ vT,
                       const float* __restrict__ rpbF, const float* __restrict__ mask,
                       f16* __restrict__ aout) {
  __shared__ f16 lP[64 * 64];
  const int win = blockIdx.x;
  const int h   = blockIdx.y;
  const int lane = threadIdx.x;
  const long rb = (long)win * N_WIN;
  const int nl = lane & 15;
  const int koff8  = (lane & 16) ? 8 : 0;
  const int koff16 = (lane & 16) ? 16 : 0;
  const int half   = (lane & 16) ? 8 : 0;

  // B operands for QK^T (B = K^T): contiguous per lane, hoisted across mt
  v16h bK[4];
#pragma unroll
  for (int nt = 0; nt < 4; ++nt)
    bK[nt] = *(const v16h*)(kkh + (rb + nt * 16 + nl) * C_DIM + h * HD + koff16);

  v8f s[4][4];
#pragma unroll
  for (int mt = 0; mt < 4; ++mt) {
    const f16* qp = qw + (rb + mt * 16 + nl) * C_DIM + h * HD + koff16;
    v16h a;
#pragma unroll
    for (int i = 0; i < 8; ++i) a[i] = qp[i];
#pragma unroll
    for (int i = 0; i < 8; ++i) a[i + 8] = qp[8 + i];
#pragma unroll
    for (int nt = 0; nt < 4; ++nt)
      s[mt][nt] = WMMA_F32_F16(a, bK[nt], v8f_zero());
  }

  // scale + rpb + mask + row softmax (shuffle reductions within lane halves)
  const float* mrow = mask + (long)(win & (NWIN - 1)) * N_WIN * N_WIN;
  const float* rh   = rpbF + (long)h * N_WIN * N_WIN;
  const float scale = 0.17677669529663687f;  // hd^-0.5
#pragma unroll
  for (int mt = 0; mt < 4; ++mt) {
#pragma unroll
    for (int j = 0; j < 8; ++j) {
      int m = mt * 16 + j + half;
      float vv[4], mx = -3.0e38f;
#pragma unroll
      for (int nt = 0; nt < 4; ++nt) {
        int n = nt * 16 + nl;
        float t = s[mt][nt][j] * scale + rh[m * 64 + n] + mrow[m * 64 + n];
        vv[nt] = t;
        mx = fmaxf(mx, t);
      }
#pragma unroll
      for (int o = 8; o >= 1; o >>= 1) mx = fmaxf(mx, __shfl_xor(mx, o, 32));
      float sum = 0.0f;
#pragma unroll
      for (int nt = 0; nt < 4; ++nt) { vv[nt] = __expf(vv[nt] - mx); sum += vv[nt]; }
#pragma unroll
      for (int o = 8; o >= 1; o >>= 1) sum += __shfl_xor(sum, o, 32);
      float inv = 1.0f / sum;
#pragma unroll
      for (int nt = 0; nt < 4; ++nt) lP[m * 64 + nt * 16 + nl] = (f16)(vv[nt] * inv);
    }
  }
  __syncthreads();

  // AV: attn(64x64) @ v(64x32); B operands from vT, contiguous per lane
  v16h bV[2][2];
#pragma unroll
  for (int kt = 0; kt < 2; ++kt)
#pragma unroll
    for (int nt = 0; nt < 2; ++nt)
      bV[kt][nt] = *(const v16h*)(vT + (((long)win * NH + h) * 32 + nt * 16 + nl) * 64
                                  + kt * 32 + koff16);
  v8f o2[4][2];
#pragma unroll
  for (int mt = 0; mt < 4; ++mt)
#pragma unroll
    for (int nt = 0; nt < 2; ++nt) o2[mt][nt] = v8f_zero();
#pragma unroll
  for (int kt = 0; kt < 2; ++kt)
#pragma unroll
    for (int mt = 0; mt < 4; ++mt) {
      v16h a = load_A32(lP + mt * 16 * 64 + kt * 32, 64, lane);
#pragma unroll
      for (int nt = 0; nt < 2; ++nt)
        o2[mt][nt] = WMMA_F32_F16(a, bV[kt][nt], o2[mt][nt]);
    }

#pragma unroll
  for (int mt = 0; mt < 4; ++mt)
#pragma unroll
    for (int nt = 0; nt < 2; ++nt)
#pragma unroll
      for (int j = 0; j < 8; ++j) {
        int m = mt * 16 + j + half;
        int d = nt * 16 + nl;
        aout[(rb + m) * C_DIM + h * HD + d] = (f16)o2[mt][nt][j];
      }
}

// ---------------------------------------------------------------------------
// K4: output projection + window reverse + un-shift + residual -> x2 (f32)
// ---------------------------------------------------------------------------
__global__ void k_gemm_proj(const f16* __restrict__ A, const f16* __restrict__ Bsw,
                            const float* __restrict__ bias, const float* __restrict__ x,
                            float* __restrict__ x2) {
  __shared__ f16 lA[2 * 64 * 32];
  v8f acc[4];
  long rowBase = (long)blockIdx.x * 64;
  int ntIdx = blockIdx.y * 4 + (threadIdx.x >> 5);
  gemm_tile_sw(A, C_DIM, rowBase, Bsw, 12, ntIdx, C_DIM, acc, lA);
  int lane = threadIdx.x & 31;
  int n = ntIdx * 16 + (lane & 15);
  float bv = bias[n];
  int mo = (lane & 16) ? 8 : 0;
#pragma unroll
  for (int mt = 0; mt < 4; ++mt)
#pragma unroll
    for (int j = 0; j < 8; ++j) {
      long r = rowBase + mt * 16 + j + mo;
      int b;
      long l = win_row_to_l(r, &b);
      long idx = ((long)b * L_TOK + l) * C_DIM + n;
      x2[idx] = x[idx] + acc[mt][j] + bv;
    }
}

// ---------------------------------------------------------------------------
// K5: LayerNorm2 over x2 -> hln (f16)
// ---------------------------------------------------------------------------
__global__ void k_ln2(const float* __restrict__ x2, const float* __restrict__ g,
                      const float* __restrict__ bb, f16* __restrict__ hln) {
  long r = (long)blockIdx.x * 8 + (threadIdx.x >> 5);
  int lane = threadIdx.x & 31;
  const float* in = x2 + r * C_DIM;
  f16* out = hln + r * C_DIM;
  float v[6], s = 0.0f;
#pragma unroll
  for (int j = 0; j < 6; ++j) { v[j] = in[lane + 32 * j]; s += v[j]; }
#pragma unroll
  for (int o = 16; o >= 1; o >>= 1) s += __shfl_xor(s, o, 32);
  float mean = s * (1.0f / C_DIM);
  float vs = 0.0f;
#pragma unroll
  for (int j = 0; j < 6; ++j) { float d = v[j] - mean; vs += d * d; }
#pragma unroll
  for (int o = 16; o >= 1; o >>= 1) vs += __shfl_xor(vs, o, 32);
  float inv = rsqrtf(vs * (1.0f / C_DIM) + 1e-5f);
#pragma unroll
  for (int j = 0; j < 6; ++j) {
    int c = lane + 32 * j;
    out[c] = (f16)((v[j] - mean) * inv * g[c] + bb[c]);
  }
}

// ---------------------------------------------------------------------------
// K6: FC1 GEMM + exact GELU -> h1 (f16)
// ---------------------------------------------------------------------------
__global__ void k_gemm_fc1(const f16* __restrict__ A, const f16* __restrict__ Bsw,
                           const float* __restrict__ bias, f16* __restrict__ h1) {
  __shared__ f16 lA[2 * 64 * 32];
  v8f acc[4];
  long rowBase = (long)blockIdx.x * 64;
  int ntIdx = blockIdx.y * 4 + (threadIdx.x >> 5);
  gemm_tile_sw(A, C_DIM, rowBase, Bsw, 48, ntIdx, C_DIM, acc, lA);
  int lane = threadIdx.x & 31;
  int n = ntIdx * 16 + (lane & 15);
  float bv = bias[n];
  int mo = (lane & 16) ? 8 : 0;
#pragma unroll
  for (int mt = 0; mt < 4; ++mt)
#pragma unroll
    for (int j = 0; j < 8; ++j) {
      long m = rowBase + mt * 16 + j + mo;
      float t = acc[mt][j] + bv;
      float ge = 0.5f * t * (1.0f + erff(t * 0.70710678118654752f));
      h1[m * HD_DIM + n] = (f16)ge;
    }
}

// ---------------------------------------------------------------------------
// K7: FC2 GEMM + bias + residual(x2) -> d_out (f32)
// ---------------------------------------------------------------------------
__global__ void k_gemm_fc2(const f16* __restrict__ A, const f16* __restrict__ Bsw,
                           const float* __restrict__ bias, const float* __restrict__ x2,
                           float* __restrict__ out) {
  __shared__ f16 lA[2 * 64 * 32];
  v8f acc[4];
  long rowBase = (long)blockIdx.x * 64;
  int ntIdx = blockIdx.y * 4 + (threadIdx.x >> 5);
  gemm_tile_sw(A, HD_DIM, rowBase, Bsw, 12, ntIdx, HD_DIM, acc, lA);
  int lane = threadIdx.x & 31;
  int n = ntIdx * 16 + (lane & 15);
  float bv = bias[n];
  int mo = (lane & 16) ? 8 : 0;
#pragma unroll
  for (int mt = 0; mt < 4; ++mt)
#pragma unroll
    for (int j = 0; j < 8; ++j) {
      long m = rowBase + mt * 16 + j + mo;
      long idx = m * C_DIM + n;
      out[idx] = x2[idx] + acc[mt][j] + bv;
    }
}

// ---------------------------------------------------------------------------
// Host launcher
// ---------------------------------------------------------------------------
extern "C" void kernel_launch(void* const* d_in, const int* in_sizes, int n_in,
                              void* d_out, int out_size, void* d_ws, size_t ws_size,
                              hipStream_t stream) {
  const float* x        = (const float*)d_in[0];
  const float* maskm    = (const float*)d_in[1];
  const float* skip     = (const float*)d_in[2];
  const float* x_up     = (const float*)d_in[3];
  const float* norm1_g  = (const float*)d_in[4];
  const float* norm1_b  = (const float*)d_in[5];
  const float* kv_w     = (const float*)d_in[6];
  const float* kv_b     = (const float*)d_in[7];
  const float* rpb_tab  = (const float*)d_in[8];
  const float* proj_w   = (const float*)d_in[9];
  const float* proj_b   = (const float*)d_in[10];
  const float* norm2_g  = (const float*)d_in[11];
  const float* norm2_b  = (const float*)d_in[12];
  const float* fc1_w    = (const float*)d_in[13];
  const float* fc1_b    = (const float*)d_in[14];
  const float* fc2_w    = (const float*)d_in[15];
  const float* fc2_b    = (const float*)d_in[16];

  char* ws = (char*)d_ws;
  // Workspace layout (bytes). h1 (100.66 MB) aliases qw+skw+kkh+vT
  // (4 x 25165824 = 100663296), all dead before FC1 runs.
  const size_t OFF_WKV   = 0;                            // 147456  B f16 (swizzled)
  const size_t OFF_WPROJ = OFF_WKV   + 147456;           // 73728
  const size_t OFF_WFC1  = OFF_WPROJ + 73728;            // 294912
  const size_t OFF_WFC2  = OFF_WFC1  + 294912;           // 294912
  const size_t OFF_RPB   = OFF_WFC2  + 294912;           // 98304   B f32
  const size_t OFF_QW    = OFF_RPB   + 98304;            // 25165824 B f16
  const size_t OFF_SKW   = OFF_QW    + 25165824;         // 25165824
  const size_t OFF_KKH   = OFF_SKW   + 25165824;         // 25165824
  const size_t OFF_VT    = OFF_KKH   + 25165824;         // 25165824
  const size_t OFF_H1    = OFF_QW;                       // alias: 100663296
  const size_t OFF_AOUT  = OFF_VT    + 25165824;         // 25165824
  const size_t OFF_X2    = OFF_AOUT  + 25165824;         // 50331648 B f32
  const size_t OFF_HLN   = OFF_X2    + 50331648;         // 25165824 B f16

  f16*   wKV   = (f16*)(ws + OFF_WKV);
  f16*   wPROJ = (f16*)(ws + OFF_WPROJ);
  f16*   wFC1  = (f16*)(ws + OFF_WFC1);
  f16*   wFC2  = (f16*)(ws + OFF_WFC2);
  float* rpbF  = (float*)(ws + OFF_RPB);
  f16*   qw    = (f16*)(ws + OFF_QW);
  f16*   skw   = (f16*)(ws + OFF_SKW);
  f16*   kkh   = (f16*)(ws + OFF_KKH);
  f16*   vT    = (f16*)(ws + OFF_VT);
  f16*   h1    = (f16*)(ws + OFF_H1);
  f16*   aout  = (f16*)(ws + OFF_AOUT);
  float* x2    = (float*)(ws + OFF_X2);
  f16*   hln   = (f16*)(ws + OFF_HLN);
  float* out   = (float*)d_out;

  // 0) weights -> f16 swizzled B-operand tiles, rpb gather
  hipLaunchKernelGGL(k_prep, dim3(576), dim3(256), 0, stream,
                     kv_w, proj_w, fc1_w, fc2_w, rpb_tab, wKV, wPROJ, wFC1, wFC2, rpbF);
  // 1) LN1 + shift + partition
  hipLaunchKernelGGL(k_ln_part, dim3(ROWS / 8), dim3(256), 0, stream,
                     skip, x_up, norm1_g, norm1_b, skw, qw);
  // 2) KV GEMM: 65536 x 384, K = 192 (K -> kkh, V -> vT transposed)
  hipLaunchKernelGGL(k_gemm_kv, dim3(ROWS / 64, (2 * C_DIM) / 64), dim3(128), 0, stream,
                     skw, wKV, kv_b, kkh, vT);
  // 3) fused attention per (window, head)
  hipLaunchKernelGGL(k_attn, dim3(B_BAT * NWIN, NH), dim3(32), 0, stream,
                     qw, kkh, vT, rpbF, maskm, aout);
  // 4) proj GEMM + window reverse + residual
  hipLaunchKernelGGL(k_gemm_proj, dim3(ROWS / 64, C_DIM / 64), dim3(128), 0, stream,
                     aout, wPROJ, proj_b, x, x2);
  // 5) LN2
  hipLaunchKernelGGL(k_ln2, dim3(ROWS / 8), dim3(256), 0, stream,
                     x2, norm2_g, norm2_b, hln);
  // 6) FC1 + GELU
  hipLaunchKernelGGL(k_gemm_fc1, dim3(ROWS / 64, HD_DIM / 64), dim3(128), 0, stream,
                     hln, wFC1, fc1_b, h1);
  // 7) FC2 + residual -> out
  hipLaunchKernelGGL(k_gemm_fc2, dim3(ROWS / 64, C_DIM / 64), dim3(128), 0, stream,
                     h1, wFC2, fc2_b, x2, out);
}
